// SchNet_73735998538386
// MI455X (gfx1250) — matrix operand order repaired and checked by reading, not compiled, not run on hardware
//
#include <hip/hip_runtime.h>
#include <hip/hip_bf16.h>

#define CUTOFF_F 10.0f
#define NUMG 50
#define LOG2F_ 0.6931471805599453f
#define PI_ 3.14159265358979323846f

typedef __attribute__((ext_vector_type(16))) _Float16 v16h;
typedef __attribute__((ext_vector_type(8)))  _Float16 v8h;
typedef __attribute__((ext_vector_type(8)))  float    v8f;
typedef __attribute__((ext_vector_type(4)))  float    v4f;

union HFrag { v16h v; v8h h[2]; _Float16 e[16]; };

__device__ __forceinline__ float sspf(float x) {
    // softplus(x) - log(2), numerically stable
    float sp = (x > 20.0f) ? x : log1pf(__expf(x));
    return sp - LOG2F_;
}

__device__ __forceinline__ v8f wmma16(const HFrag& a, const HFrag& b, v8f c) {
    return __builtin_amdgcn_wmma_f32_16x16x32_f16(
        /*neg_a=*/false, a.v, /*neg_b=*/false, b.v,
        /*c_mod=*/(short)0, c, /*reuse_a=*/false, /*reuse_b=*/false);
}

// Load a 16-half A fragment from a row of f32: two 8-float runs at
// [base, base+8) and [base+16, base+24), via four 16B vector loads.
__device__ __forceinline__ void load_afrag_f32(HFrag& a, const float* base) {
    v4f x0 = *(const v4f*)(base);
    v4f x1 = *(const v4f*)(base + 4);
    v4f y0 = *(const v4f*)(base + 16);
    v4f y1 = *(const v4f*)(base + 20);
#pragma unroll
    for (int j = 0; j < 4; ++j) {
        a.e[j]      = (_Float16)x0[j];
        a.e[4 + j]  = (_Float16)x1[j];
        a.e[8 + j]  = (_Float16)y0[j];
        a.e[12 + j] = (_Float16)y1[j];
    }
}

// ---------------------------------------------------------------- utilities

__global__ void zero_k(float* __restrict__ p, long n) {
    long i = (long)blockIdx.x * blockDim.x + threadIdx.x;
    long s = (long)gridDim.x * blockDim.x;
    for (; i < n; i += s) p[i] = 0.0f;
}

__global__ void embed_k(const int* __restrict__ x, const float* __restrict__ emb,
                        float* __restrict__ h, int N) {
    long tot = (long)N * 128;
    long i = (long)blockIdx.x * blockDim.x + threadIdx.x;
    long s = (long)gridDim.x * blockDim.x;
    for (; i < tot; i += s) {
        int n = (int)(i >> 7), f = (int)(i & 127);
        h[i] = emb[(size_t)x[n] * 128 + f];
    }
}

// gaussian smearing -> f16 [E,64] (padded) and cosine cutoff envelope [E]
__global__ void smear_k(const float* __restrict__ dist, _Float16* __restrict__ ea,
                        float* __restrict__ cenv, int E) {
    int e = blockIdx.x * blockDim.x + threadIdx.x;
    if (e >= E) return;
    float d = dist[e];
    cenv[e] = 0.5f * (__cosf(d * (PI_ / CUTOFF_F)) + 1.0f);
    const float step  = CUTOFF_F / (float)(NUMG - 1);
    const float coeff = -0.5f / (step * step);
    _Float16* row = ea + (size_t)e * 64;
#pragma unroll
    for (int g = 0; g < 64; ++g) {
        float v = 0.0f;
        if (g < NUMG) { float t = d - (float)g * step; v = __expf(coeff * t * t); }
        row[g] = (_Float16)v;
    }
}

// Pack a row-major [K,F] f32 weight into WMMA B-fragment order (f16).
// frag(kc,nc): 32 lanes x 16 halfs; lane L holds col nc*16+(L&15),
// K = kc*32 + ((L>=16)?8:0) + ((j>=8)?16:0) + (j&7)  for j in 0..15. Rows >= srcK are zero-padded.
__global__ void pack_w(const float* __restrict__ W, _Float16* __restrict__ out,
                       int K, int F, int srcK) {
    int idx = blockIdx.x * blockDim.x + threadIdx.x;
    if (idx >= K * F) return;
    int j    = idx & 15;
    int lane = (idx >> 4) & 31;
    int f    = idx >> 9;
    int nper = F >> 4;
    int nc = f % nper, kc = f / nper;
    int krow = kc * 32 + (j & 8) * 2 + ((lane >> 4) & 1) * 8 + (j & 7);
    int col  = nc * 16 + (lane & 15);
    float v = (krow < srcK) ? W[(size_t)krow * F + col] : 0.0f;
    out[idx] = (_Float16)v;
}

// ------------------------------------------------------- node-side GEMM
// Out[M,FOUT] = op( A[M,KDIM] @ Bp + bias );  8 waves x 16 rows per block.
template <int KDIM, int FOUT, bool BIAS, bool SSP, bool ADD>
__global__ __launch_bounds__(256) void node_gemm_k(
    const float* __restrict__ A, const _Float16* __restrict__ Bp,
    const float* __restrict__ bias, float* __restrict__ Out, int M) {
    constexpr int KC = KDIM / 32, NC = FOUT / 16;
    const int lane = threadIdx.x & 31;
    const int wv   = threadIdx.x >> 5;
    const int hi   = lane >> 4;
    const int m    = lane & 15;
    const int rowA = blockIdx.x * 128 + wv * 16 + m;
    const int rA   = (rowA < M) ? rowA : (M - 1);
    const float* arow = A + (size_t)rA * KDIM;
    const int kb = hi * 8;

    HFrag a[KC];
#pragma unroll
    for (int kc = 0; kc < KC; ++kc)
        load_afrag_f32(a[kc], arow + kc * 32 + kb);

    // hoisted output row pointers for this lane's 8 result rows
    const int rbase = blockIdx.x * 128 + wv * 16 + hi * 8;
    float* op[8];
    bool   ok[8];
#pragma unroll
    for (int r = 0; r < 8; ++r) {
        int rr = rbase + r;
        ok[r] = (rr < M);
        op[r] = Out + (size_t)(ok[r] ? rr : 0) * FOUT;
    }

#pragma unroll
    for (int nc = 0; nc < NC; ++nc) {
        float bval = BIAS ? bias[nc * 16 + m] : 0.0f;
        v8f acc;
#pragma unroll
        for (int i = 0; i < 8; ++i) acc[i] = bval;
#pragma unroll
        for (int kc = 0; kc < KC; ++kc) {
            HFrag b;
            const v8h* bp = (const v8h*)(Bp + ((size_t)(kc * NC + nc) * 32 + lane) * 16);
            b.h[0] = bp[0]; b.h[1] = bp[1];
            acc = wmma16(a[kc], b, acc);
        }
        const int n = nc * 16 + m;
#pragma unroll
        for (int r = 0; r < 8; ++r) {
            if (ok[r]) {
                float v = acc[r];
                if (SSP) v = sspf(v);
                if (ADD) op[r][n] += v;
                else     op[r][n]  = v;
            }
        }
    }
}

// ------------------------------------------------------- fused edge kernel
// Per 16-edge tile (one wave):
//   T1 = ssp(ea @ w1 + b1)           [16,128]   (wmma, ssp, staged in LDS)
//   W  = (T1 @ w2 + b2) * C[e]       [16,128]   (wmma)
//   agg[dst[e]] += W * xf[src[e]]               (L2 atomics)
__global__ __launch_bounds__(256) void edge_k(
    const _Float16* __restrict__ ea, const float* __restrict__ cenv,
    const float* __restrict__ xf,
    const int* __restrict__ src, const int* __restrict__ dst,
    const _Float16* __restrict__ w1p, const float* __restrict__ b1,
    const _Float16* __restrict__ w2p, const float* __restrict__ b2,
    float* __restrict__ agg, int E) {
    __shared__ __align__(16) _Float16 ldsT1[8][16 * 128];
    const int lane = threadIdx.x & 31;
    const int wv   = threadIdx.x >> 5;
    const int hi   = lane >> 4;
    const int m    = lane & 15;
    const int kb   = hi * 8;
    const int ntiles = E >> 4;
    const int tstep  = gridDim.x * 8;

    for (int tile = blockIdx.x * 8 + wv; tile < ntiles; tile += tstep) {
        const int e0 = tile * 16;

        // prefetch next tile's gaussian rows (global_prefetch_b8)
        {
            int tn = tile + tstep;
            if (tn < ntiles)
                __builtin_prefetch(ea + (size_t)(tn * 16 + m) * 64, 0, 1);
        }

        // A1 fragments from gaussian features (K = 64 -> 2 chunks)
        HFrag a1[2];
        const _Float16* arow = ea + (size_t)(e0 + m) * 64;
#pragma unroll
        for (int kc = 0; kc < 2; ++kc) {
            a1[kc].h[0] = *(const v8h*)(arow + kc * 32 + kb);
            a1[kc].h[1] = *(const v8h*)(arow + kc * 32 + kb + 16);
        }

        // GEMM1 + shifted-softplus -> LDS (C-layout scatter into row-major tile)
#pragma unroll
        for (int nc = 0; nc < 8; ++nc) {
            float bval = b1[nc * 16 + m];
            v8f acc;
#pragma unroll
            for (int i = 0; i < 8; ++i) acc[i] = bval;
#pragma unroll
            for (int kc = 0; kc < 2; ++kc) {
                HFrag b;
                const v8h* bp = (const v8h*)(w1p + ((size_t)(kc * 8 + nc) * 32 + lane) * 16);
                b.h[0] = bp[0]; b.h[1] = bp[1];
                acc = wmma16(a1[kc], b, acc);
            }
#pragma unroll
            for (int r = 0; r < 8; ++r)
                ldsT1[wv][(r + hi * 8) * 128 + nc * 16 + m] = (_Float16)sspf(acc[r]);
        }

        // wave-local LDS handoff (DS ops are in-order; block compiler reordering)
        asm volatile("s_wait_dscnt 0" ::: "memory");

        // A2 fragments for GEMM2 (K = 128 -> 4 chunks), row-major LDS reads
        HFrag a2[4];
        const _Float16* trow = &ldsT1[wv][m * 128];
#pragma unroll
        for (int kc = 0; kc < 4; ++kc) {
            a2[kc].h[0] = *(const v8h*)(trow + kc * 32 + kb);
            a2[kc].h[1] = *(const v8h*)(trow + kc * 32 + kb + 16);
        }
        asm volatile("" ::: "memory");

        // per-edge scalars + hoisted row base pointers for this lane's 8 rows
        float cv[8];
        const float* xp[8];
        float* ap[8];
#pragma unroll
        for (int r = 0; r < 8; ++r) {
            int e = e0 + hi * 8 + r;
            cv[r] = cenv[e];
            xp[r] = xf  + (size_t)src[e] * 128;
            ap[r] = agg + (size_t)dst[e] * 128;
        }

        // GEMM2 + cutoff + gather-multiply + scatter-add
#pragma unroll
        for (int nc = 0; nc < 8; ++nc) {
            float bval = b2[nc * 16 + m];
            v8f acc;
#pragma unroll
            for (int i = 0; i < 8; ++i) acc[i] = bval;
#pragma unroll
            for (int kc = 0; kc < 4; ++kc) {
                HFrag b;
                const v8h* bp = (const v8h*)(w2p + ((size_t)(kc * 8 + nc) * 32 + lane) * 16);
                b.h[0] = bp[0]; b.h[1] = bp[1];
                acc = wmma16(a2[kc], b, acc);
            }
            const int n = nc * 16 + m;
#pragma unroll
            for (int r = 0; r < 8; ++r) {
                float w = acc[r] * cv[r];
                atomicAdd(ap[r] + n, w * xp[r][n]);
            }
        }
    }
}

// ------------------------------------------------------- output head

__global__ void colsum_k(const float* __restrict__ t2, float* __restrict__ hagg, int N) {
    int c       = threadIdx.x & 63;
    int rstart  = (int)((blockIdx.x * blockDim.x + threadIdx.x) >> 6);
    int rstride = (int)((gridDim.x * blockDim.x) >> 6);
    float s = 0.0f;
    for (int r = rstart; r < N; r += rstride) s += t2[(size_t)r * 64 + c];
    atomicAdd(&hagg[c], s);
}

__global__ void final_k(const float* __restrict__ hagg, const float* __restrict__ row,
                        const float* __restrict__ rob, float* __restrict__ out) {
    int j = threadIdx.x;
    if (j < 12) {
        float s = rob[j];
#pragma unroll
        for (int k = 0; k < 64; ++k) s += hagg[k] * row[k * 12 + j];
        out[j] = s;
    }
}

// ------------------------------------------------------- host driver

extern "C" void kernel_launch(void* const* d_in, const int* in_sizes, int n_in,
                              void* d_out, int out_size, void* d_ws, size_t ws_size,
                              hipStream_t stream) {
    (void)n_in; (void)out_size; (void)ws_size;
    const int*   x       = (const int*)d_in[0];
    const int*   eidx    = (const int*)d_in[1];
    const float* edist   = (const float*)d_in[2];
    const float* emb     = (const float*)d_in[3];
    const float* mlp_w1  = (const float*)d_in[4];
    const float* mlp_b1  = (const float*)d_in[5];
    const float* mlp_w2  = (const float*)d_in[6];
    const float* mlp_b2  = (const float*)d_in[7];
    const float* lin1_w  = (const float*)d_in[8];
    const float* lin2_w  = (const float*)d_in[9];
    const float* lin2_b  = (const float*)d_in[10];
    const float* lin_w   = (const float*)d_in[11];
    const float* lin_b   = (const float*)d_in[12];
    const float* out1_w  = (const float*)d_in[13];
    const float* out1_b  = (const float*)d_in[14];
    const float* out2_w  = (const float*)d_in[15];
    const float* out2_b  = (const float*)d_in[16];
    const float* ro_w    = (const float*)d_in[17];
    const float* ro_b    = (const float*)d_in[18];

    const int N = in_sizes[0];
    const int E = in_sizes[2];
    const int L = in_sizes[5] / 128;   // mlp_b1 is [L,128]
    const int* src  = eidx;
    const int* dstp = eidx + E;

    char* ws = (char*)d_ws;
    size_t off = 0;
    auto take = [&](size_t bytes) -> char* {
        off = (off + 255) & ~(size_t)255;
        char* p = ws + off; off += bytes; return p;
    };
    float*    h    = (float*)take((size_t)N * 128 * 4);
    float*    xf   = (float*)take((size_t)N * 128 * 4);
    float*    agg  = (float*)take((size_t)N * 128 * 4);
    float*    tmp  = (float*)take((size_t)N * 128 * 4);
    _Float16* eah  = (_Float16*)take((size_t)E * 64 * 2);
    float*    cenv = (float*)take((size_t)E * 4);
    float*    hagg = (float*)take(64 * 4);
    _Float16* w1p  = (_Float16*)take((size_t)L * 64 * 128 * 2);
    _Float16* w2p  = (_Float16*)take((size_t)L * 128 * 128 * 2);
    _Float16* l1p  = (_Float16*)take((size_t)L * 128 * 128 * 2);
    _Float16* l2p  = (_Float16*)take((size_t)L * 128 * 128 * 2);
    _Float16* lnp  = (_Float16*)take((size_t)L * 128 * 128 * 2);
    _Float16* o1p  = (_Float16*)take((size_t)128 * 64 * 2);
    _Float16* o2p  = (_Float16*)take((size_t)64 * 64 * 2);

    // Pack all weights into WMMA fragment order (f16), once.
    for (int l = 0; l < L; ++l) {
        pack_w<<<(64 * 128) / 256, 256, 0, stream>>>(mlp_w1 + (size_t)l * NUMG * 128,
                                                     w1p + (size_t)l * 64 * 128, 64, 128, NUMG);
        pack_w<<<(128 * 128) / 256, 256, 0, stream>>>(mlp_w2 + (size_t)l * 128 * 128,
                                                      w2p + (size_t)l * 128 * 128, 128, 128, 128);
        pack_w<<<(128 * 128) / 256, 256, 0, stream>>>(lin1_w + (size_t)l * 128 * 128,
                                                      l1p + (size_t)l * 128 * 128, 128, 128, 128);
        pack_w<<<(128 * 128) / 256, 256, 0, stream>>>(lin2_w + (size_t)l * 128 * 128,
                                                      l2p + (size_t)l * 128 * 128, 128, 128, 128);
        pack_w<<<(128 * 128) / 256, 256, 0, stream>>>(lin_w + (size_t)l * 128 * 128,
                                                      lnp + (size_t)l * 128 * 128, 128, 128, 128);
    }
    pack_w<<<(128 * 64) / 256, 256, 0, stream>>>(out1_w, o1p, 128, 64, 128);
    pack_w<<<(64 * 64) / 256, 256, 0, stream>>>(out2_w, o2p, 64, 64, 64);

    embed_k<<<1024, 256, 0, stream>>>(x, emb, h, N);
    smear_k<<<(E + 255) / 256, 256, 0, stream>>>(edist, eah, cenv, E);

    const int gb = (N + 127) / 128;   // node-GEMM blocks (128 rows each)
    const int eb = E / 256;           // edge blocks: 8 waves, 2 tiles/wave

    for (int l = 0; l < L; ++l) {
        node_gemm_k<128, 128, false, false, false><<<gb, 256, 0, stream>>>(
            h, l1p + (size_t)l * 128 * 128, nullptr, xf, N);
        zero_k<<<2048, 256, 0, stream>>>(agg, (long)N * 128);
        edge_k<<<eb, 256, 0, stream>>>(eah, cenv, xf, src, dstp,
                                       w1p + (size_t)l * 64 * 128, mlp_b1 + (size_t)l * 128,
                                       w2p + (size_t)l * 128 * 128, mlp_b2 + (size_t)l * 128,
                                       agg, E);
        node_gemm_k<128, 128, true, true, false><<<gb, 256, 0, stream>>>(
            agg, l2p + (size_t)l * 128 * 128, lin2_b + (size_t)l * 128, tmp, N);
        node_gemm_k<128, 128, true, false, true><<<gb, 256, 0, stream>>>(
            tmp, lnp + (size_t)l * 128 * 128, lin_b + (size_t)l * 128, h, N);
    }

    // output head: t = ssp(h@out1+b); t2 = t@out2+b; colsum; @ro_w+ro_b
    node_gemm_k<128, 64, true, true, false><<<gb, 256, 0, stream>>>(h, o1p, out1_b, tmp, N);
    node_gemm_k<64, 64, true, false, false><<<gb, 256, 0, stream>>>(tmp, o2p, out2_b, xf, N);
    zero_k<<<1, 64, 0, stream>>>(hagg, 64);
    colsum_k<<<64, 256, 0, stream>>>(xf, hagg, N);
    final_k<<<1, 32, 0, stream>>>(hagg, ro_w, ro_b, (float*)d_out);
}
